// LinearSelfAttention_86062554677353
// MI455X (gfx1250) — matrix operand order repaired and checked by reading, not compiled, not run on hardware
//
#include <hip/hip_runtime.h>
#include <hip/hip_bf16.h>
#include <math.h>

// ---------------------------------------------------------------------------
// Linear self-attention, CDNA5 (gfx1250) WMMA bf16 implementation.
// B=4, S=4096, C=1024, H=16, D=64.
// ---------------------------------------------------------------------------

typedef __bf16 bf16;
typedef __attribute__((ext_vector_type(16))) __bf16 bf16x16;
typedef __attribute__((ext_vector_type(8)))  __bf16 bf16x8;
typedef __attribute__((ext_vector_type(8)))  float   v8f;
typedef int v4i __attribute__((vector_size(16)));   // matches builtin's V4i

#define BB 4
#define SS 4096
#define CC 1024
#define HH 16
#define DD 64
#define MM (BB * SS)   // 16384 rows in the big GEMMs

// LDS row stride (elements) for staged B tiles: 40 (=80B) to spread banks.
#define BSTRIDE 40

__device__ __forceinline__ v8f wmma_bf16(bf16x16 a, bf16x16 b, v8f c) {
  return __builtin_amdgcn_wmma_f32_16x16x32_bf16(
      /*neg_a=*/false, a, /*neg_b=*/false, b,
      /*c_mod=*/(short)0, c, /*reuse_a=*/false, /*reuse_b=*/false);
}

union ATile { bf16x16 v; bf16x8 h[2]; };

// ---------------------------------------------------------------------------
// CDNA5 async global->LDS copy (ASYNCcnt path), with a safe fallback.
// Builtin signature (confirmed by round-2 diagnostic):
//   void __builtin_amdgcn_global_load_async_to_lds_b128(v4i AS1*, v4i AS3*, imm, imm)
// ---------------------------------------------------------------------------
#if defined(__has_builtin)
# if __has_builtin(__builtin_amdgcn_global_load_async_to_lds_b128)
#  define HAVE_ASYNC_LDS 1
# endif
#endif

#ifdef HAVE_ASYNC_LDS
# define ASYNC_COPY_B128(ldsptr, gptr)                                         \
    __builtin_amdgcn_global_load_async_to_lds_b128(                            \
        (__attribute__((address_space(1))) v4i*)(gptr),                        \
        (__attribute__((address_space(3))) v4i*)(ldsptr), 0, 0)
# if __has_builtin(__builtin_amdgcn_s_wait_asynccnt)
#  define ASYNC_WAIT1() __builtin_amdgcn_s_wait_asynccnt(1)
# else
#  define ASYNC_WAIT1() asm volatile("s_wait_asynccnt 0x1" ::: "memory")
# endif
#else
// Fallback: stage through registers + ds_store; __syncthreads() (DScnt) orders it.
# define ASYNC_COPY_B128(ldsptr, gptr)                                         \
    do { *(bf16x8*)(ldsptr) = *(const bf16x8*)(gptr); } while (0)
# define ASYNC_WAIT1() do {} while (0)
#endif

// ---------------------------------------------------------------------------
// Conversions
// ---------------------------------------------------------------------------
__global__ void cvt_x_kernel(const float* __restrict__ x, bf16* __restrict__ out, int n) {
  int i = blockIdx.x * blockDim.x + threadIdx.x;
  if (i >= n) return;
  float v = x[i];
  if (!__builtin_isfinite(v)) v = 0.0f;
  v = fminf(fmaxf(v, -10000.0f), 10000.0f);
  out[i] = (bf16)v;
}

__global__ void cvt_f32_bf16(const float* __restrict__ in, bf16* __restrict__ out, int n) {
  int i = blockIdx.x * blockDim.x + threadIdx.x;
  if (i >= n) return;
  out[i] = (bf16)in[i];
}

__global__ void zero_f32(float* __restrict__ p, int n) {
  int i = blockIdx.x * blockDim.x + threadIdx.x;
  if (i < n) p[i] = 0.0f;
}

// ---------------------------------------------------------------------------
// Big GEMM: Y[m,n] = sum_k A[m,k] * W[n,k] + bias[n]   (M=16384,N=1024,K=1024)
// Block tile: 256 rows x 64 cols. 8 waves share the 64 columns -> the B chunk
// (64 n-rows x 32 k = 4KB) is staged once per K-step into LDS, double-buffered
// via async global->LDS copies. Each wave computes a 32x64 block (8 WMMA per
// K-step) with its A fragments software-pipelined in registers.
// MODE 0: phi(y) -> qf   (B,H,S,D)
// MODE 1: phi(y) -> kfT  (B,H,D,S)
// MODE 2:      y -> vT   (B,H,D,S)
// MODE 3: finite(y) -> f32 out, plain (M,N) row-major (final Wo projection)
// ---------------------------------------------------------------------------
template <int MODE>
__global__ void gemm_proj_kernel(const bf16* __restrict__ A, const bf16* __restrict__ W,
                                 const float* __restrict__ bias, void* __restrict__ outp) {
  __shared__ bf16 sB[2][64 * BSTRIDE];   // 2 x 5120B (+pad) double buffer

  const int t    = threadIdx.x;
  const int lane = t & 31;
  const int wv   = t >> 5;
  const int mband = blockIdx.x >> 4;      // 0..63  (256-row band)
  const int nblk  = blockIdx.x & 15;      // 0..15  (64-col band, shared by block)
  const int m0 = mband * 256 + wv * 32;
  const int n0 = nblk * 64;
  const int lr = lane & 15, lh = lane >> 4;

  // Per-thread async-copy job: 16B of the 64x32 B chunk.
  const int crow = t >> 2;                // 0..63 : n-row within chunk
  const int cq   = t & 3;                 // 0..3  : 8-element k-quarter
  const bf16* gB = W + (size_t)(n0 + crow) * CC + cq * 8;
  bf16* lB0 = &sB[0][crow * BSTRIDE + cq * 8];
  bf16* lB1 = &sB[1][crow * BSTRIDE + cq * 8];

  v8f acc[2][4] = {};
  const bf16* a0p = A + (size_t)(m0 + lr) * CC + lh * 16;
  const bf16* a1p = a0p + (size_t)16 * CC;

  ATile Aa0, Aa1, Ab0, Ab1;

#define LOAD_A(T0, T1, KB)                                                     \
  do {                                                                         \
    T0.h[0] = *(const bf16x8*)(a0p + (KB));                                    \
    T0.h[1] = *(const bf16x8*)(a0p + (KB) + 8);                                \
    T1.h[0] = *(const bf16x8*)(a1p + (KB));                                    \
    T1.h[1] = *(const bf16x8*)(a1p + (KB) + 8);                                \
  } while (0)

#define GEMM_STEP(CUR0, CUR1, NXT0, NXT1, LNXT, BUF, KB)                       \
  do {                                                                         \
    const int kn = ((KB) + 32 < CC) ? (KB) + 32 : 0; /* wrap = harmless dummy */\
    ASYNC_COPY_B128(LNXT, gB + kn);              /* stage next B chunk     */  \
    LOAD_A(NXT0, NXT1, kn);                      /* prefetch next A frags  */  \
    ASYNC_WAIT1();                               /* current B chunk landed */  \
    __syncthreads();                                                           \
    _Pragma("unroll")                                                          \
    for (int j = 0; j < 4; ++j) {                                              \
      ATile bt;                                                                \
      const bf16* bp = &sB[BUF][(j * 16 + lr) * BSTRIDE + lh * 16];            \
      bt.h[0] = *(const bf16x8*)(bp);                                          \
      bt.h[1] = *(const bf16x8*)(bp + 8);                                      \
      acc[0][j] = wmma_bf16(CUR0.v, bt.v, acc[0][j]);                          \
      acc[1][j] = wmma_bf16(CUR1.v, bt.v, acc[1][j]);                          \
    }                                                                          \
    __syncthreads();                                                           \
  } while (0)

  // Prologue: stage chunk kb=0 into buffer 0, preload first A fragments.
  ASYNC_COPY_B128(lB0, gB);
  LOAD_A(Aa0, Aa1, 0);

  for (int kb = 0; kb < CC; kb += 64) {
    GEMM_STEP(Aa0, Aa1, Ab0, Ab1, lB1, 0, kb);        // consume buf0, fill buf1
    GEMM_STEP(Ab0, Ab1, Aa0, Aa1, lB0, 1, kb + 32);   // consume buf1, fill buf0
  }
#undef GEMM_STEP
#undef LOAD_A

#pragma unroll
  for (int j = 0; j < 4; ++j) {
    const int n = n0 + j * 16 + lr;
    const float bval = bias[n];
    const int h = n >> 6, d = n & 63;
#pragma unroll
    for (int i = 0; i < 2; ++i) {
#pragma unroll
      for (int v = 0; v < 8; ++v) {
        const int m = m0 + i * 16 + lh * 8 + v;
        float val = acc[i][j][v] + bval;
        if (MODE <= 1) val = (val > 0.0f) ? (val + 1.0f) : __expf(val);  // phi = elu+1
        if (MODE == 3) {
          if (!__builtin_isfinite(val)) val = 0.0f;
          ((float*)outp)[(size_t)m * CC + n] = val;
        } else {
          const int b = m >> 12;            // m / S
          const int s = m & (SS - 1);       // m % S
          size_t idx;
          if (MODE == 0) idx = (((size_t)(b * HH + h)) * SS + s) * DD + d;  // (B,H,S,D)
          else           idx = (((size_t)(b * HH + h)) * DD + d) * SS + s;  // (B,H,D,S)
          ((bf16*)outp)[idx] = (bf16)val;
        }
      }
    }
  }
}

// ---------------------------------------------------------------------------
// kv GEMM: per head, kvT[e][d] = sum_s kf[s][d] * v[s][e]
// M=N=64, K=4096; 4-way K-split, f32 atomicAdd epilogue into kvT.
// ---------------------------------------------------------------------------
__global__ void gemm_kv_kernel(const bf16* __restrict__ kfT, const bf16* __restrict__ vT,
                               float* __restrict__ kvT) {
  const int lane = threadIdx.x & 31;
  const int wv   = threadIdx.x >> 5;
  const int task = blockIdx.x * 8 + wv;   // 1024 tasks = 64 heads x 4 ntiles x 4 kchunks
  const int bh = task >> 4;
  const int nt = (task >> 2) & 3;
  const int kc = task & 3;
  const int lr = lane & 15, lh = lane >> 4;

  const bf16* Ab = kfT + (size_t)bh * DD * SS;
  const bf16* Bb = vT + (size_t)bh * DD * SS + (size_t)(nt * 16 + lr) * SS + lh * 16;

  v8f acc[4] = {};
  const int k0 = kc * 1024;
  for (int kb = k0; kb < k0 + 1024; kb += 32) {
    bf16x16 bt = *(const bf16x16*)(Bb + kb);
#pragma unroll
    for (int i = 0; i < 4; ++i) {
      ATile a;
      const bf16* ap = Ab + (size_t)(i * 16 + lr) * SS + kb + lh * 16;
      a.h[0] = *(const bf16x8*)(ap);
      a.h[1] = *(const bf16x8*)(ap + 8);
      acc[i] = wmma_bf16(a.v, bt, acc[i]);
    }
  }
  float* kvp = kvT + (size_t)bh * DD * DD + (size_t)(nt * 16 + lr) * DD;  // row e
#pragma unroll
  for (int i = 0; i < 4; ++i)
#pragma unroll
    for (int v = 0; v < 8; ++v)
      atomicAdd(&kvp[i * 16 + lh * 8 + v], acc[i][v]);
}

// kf_sum[b][h][d] = sum_s kfT[b][h][d][s]
__global__ void kfsum_kernel(const bf16* __restrict__ kfT, float* __restrict__ kfsum) {
  __shared__ float red[256];
  const int row = blockIdx.x;  // B*H*D rows of length S
  const bf16* p = kfT + (size_t)row * SS;
  float s = 0.0f;
  for (int i = threadIdx.x; i < SS; i += 256) s += (float)p[i];
  red[threadIdx.x] = s;
  __syncthreads();
  for (int st = 128; st > 0; st >>= 1) {
    if (threadIdx.x < st) red[threadIdx.x] += red[threadIdx.x + st];
    __syncthreads();
  }
  if (threadIdx.x == 0) kfsum[row] = red[0];
}

// ---------------------------------------------------------------------------
// attn GEMM: out[s][e] = (qf[s][:] . kv[:][e]) / max(qf[s][:].kf_sum, 1e-6)
// Per wave: one 16-row M-tile x full D (4 N-tiles). K=64 -> 2 K-steps.
// z is computed from the already-loaded A fragments + wave shuffles.
// ---------------------------------------------------------------------------
__global__ void gemm_attn_kernel(const bf16* __restrict__ qf, const bf16* __restrict__ kvbf,
                                 const float* __restrict__ kfsum, bf16* __restrict__ outA) {
  const int lane = threadIdx.x & 31;
  const int wv   = threadIdx.x >> 5;
  const int task = blockIdx.x * 8 + wv;   // 16384 = 64 heads x 256 m-tiles
  const int bh = task >> 8;
  const int mt = task & 255;
  const int lr = lane & 15, lh = lane >> 4;

  const bf16* Ap = qf + ((size_t)bh * SS + mt * 16 + lr) * DD + lh * 16;
  const bf16* Bb = kvbf + (size_t)bh * DD * DD;            // kvT[e][d], row-major
  const float* kfs = kfsum + bh * DD;

  v8f acc[4] = {};
  float zp = 0.0f;
#pragma unroll
  for (int kb = 0; kb < DD; kb += 32) {
    ATile a;
    a.h[0] = *(const bf16x8*)(Ap + kb);
    a.h[1] = *(const bf16x8*)(Ap + kb + 8);
#pragma unroll
    for (int t = 0; t < 8; ++t) {
      zp += (float)a.v[t]     * kfs[kb + lh * 16 + t];
      zp += (float)a.v[8 + t] * kfs[kb + lh * 16 + 8 + t];
    }
#pragma unroll
    for (int j = 0; j < 4; ++j) {
      bf16x16 bt = *(const bf16x16*)(Bb + (size_t)(j * 16 + lr) * DD + kb + lh * 16);
      acc[j] = wmma_bf16(a.v, bt, acc[j]);
    }
  }
  zp += __shfl_xor(zp, 16, 32);
  float zr[8];
#pragma unroll
  for (int v = 0; v < 8; ++v) zr[v] = __shfl(zp, lh * 8 + v, 32);

  const int b = bh >> 4, h = bh & 15;
  const int s0 = mt * 16;
#pragma unroll
  for (int j = 0; j < 4; ++j) {
    const int e = j * 16 + lr;
#pragma unroll
    for (int v = 0; v < 8; ++v) {
      const int s = s0 + lh * 8 + v;
      const float z = fmaxf(zr[v], 1e-6f);
      const float val = acc[j][v] / z;
      outA[((size_t)(b * SS + s)) * CC + h * DD + e] = (bf16)val;  // (B,S,C)
    }
  }
}

// ---------------------------------------------------------------------------
// Host launcher
// ---------------------------------------------------------------------------
extern "C" void kernel_launch(void* const* d_in, const int* in_sizes, int n_in,
                              void* d_out, int out_size, void* d_ws, size_t ws_size,
                              hipStream_t stream) {
  (void)in_sizes; (void)n_in; (void)out_size; (void)ws_size;
  const float* x  = (const float*)d_in[0];
  const float* Wq = (const float*)d_in[1];
  const float* bq = (const float*)d_in[2];
  const float* Wk = (const float*)d_in[3];
  const float* bk = (const float*)d_in[4];
  const float* Wv = (const float*)d_in[5];
  const float* bv = (const float*)d_in[6];
  const float* Wo = (const float*)d_in[7];
  const float* bo = (const float*)d_in[8];

  char* ws = (char*)d_ws;
  size_t off = 0;
  auto walloc = [&](size_t bytes) -> void* {
    void* p = ws + off;
    off += (bytes + 255) & ~(size_t)255;
    return p;
  };
  // ~144 MB total workspace
  bf16*  xbf = (bf16*)walloc((size_t)MM * CC * 2);   // x in bf16; reused as out_attn later
  bf16*  wqb = (bf16*)walloc((size_t)CC * CC * 2);
  bf16*  wkb = (bf16*)walloc((size_t)CC * CC * 2);
  bf16*  wvb = (bf16*)walloc((size_t)CC * CC * 2);
  bf16*  wob = (bf16*)walloc((size_t)CC * CC * 2);
  bf16*  qf  = (bf16*)walloc((size_t)MM * CC * 2);   // (B,H,S,D)
  bf16*  kfT = (bf16*)walloc((size_t)MM * CC * 2);   // (B,H,D,S)
  bf16*  vT  = (bf16*)walloc((size_t)MM * CC * 2);   // (B,H,D,S)
  float* kvf = (float*)walloc((size_t)BB * HH * DD * DD * 4);
  bf16*  kvb = (bf16*)walloc((size_t)BB * HH * DD * DD * 2);
  float* kfs = (float*)walloc((size_t)BB * HH * DD * 4);

  // 1. convert inputs to bf16 (x: clamp NaN/Inf -> 0, clip +-1e4)
  const int NE_X = MM * CC;   // 16,777,216
  cvt_x_kernel<<<NE_X / 256, 256, 0, stream>>>(x, xbf, NE_X);
  const int NE_W = CC * CC;   // 1,048,576
  cvt_f32_bf16<<<NE_W / 256, 256, 0, stream>>>(Wq, wqb, NE_W);
  cvt_f32_bf16<<<NE_W / 256, 256, 0, stream>>>(Wk, wkb, NE_W);
  cvt_f32_bf16<<<NE_W / 256, 256, 0, stream>>>(Wv, wvb, NE_W);
  cvt_f32_bf16<<<NE_W / 256, 256, 0, stream>>>(Wo, wob, NE_W);

  // 2. Q/K/V projections (WMMA + async-LDS B staging), phi fused,
  //    K/V stored transposed
  gemm_proj_kernel<0><<<1024, 256, 0, stream>>>(xbf, wqb, bq, qf);
  gemm_proj_kernel<1><<<1024, 256, 0, stream>>>(xbf, wkb, bk, kfT);
  gemm_proj_kernel<2><<<1024, 256, 0, stream>>>(xbf, wvb, bv, vT);

  // 3. kv = kf^T v per head (WMMA, K-split atomic), kf_sum reduction
  const int NKV = BB * HH * DD * DD;  // 262,144
  zero_f32<<<(NKV + 255) / 256, 256, 0, stream>>>(kvf, NKV);
  kfsum_kernel<<<BB * HH * DD, 256, 0, stream>>>(kfT, kfs);
  gemm_kv_kernel<<<128, 256, 0, stream>>>(kfT, vT, kvf);
  cvt_f32_bf16<<<(NKV + 255) / 256, 256, 0, stream>>>(kvf, kvb, NKV);

  // 4. out = (qf kv) / z  (WMMA + in-wave z), written (B,S,C) bf16.
  //    xbf is dead now -> reuse as out_attn.
  bf16* outA = xbf;
  gemm_attn_kernel<<<2048, 256, 0, stream>>>(qf, kvb, kfs, outA);

  // 5. final projection with Wo (WMMA), finite->0, fp32 output
  gemm_proj_kernel<3><<<1024, 256, 0, stream>>>(outA, wob, bo, d_out);
}
